// CpxRBM_30769145709145
// MI455X (gfx1250) — matrix operand order repaired and checked by reading, not compiled, not run on hardware
//
#include <hip/hip_runtime.h>

typedef __attribute__((ext_vector_type(16))) _Float16 v16h;
typedef __attribute__((ext_vector_type(8)))  _Float16 v8h;
typedef __attribute__((ext_vector_type(8)))  float    v8f;
typedef __attribute__((ext_vector_type(4)))  float    v4f;

#define VIS_N 4096
#define INP_N 4096
#define OUP_N 1024
#define S_TOTAL (VIS_N - 1)        // 4095 valid circular shifts
#define NBLK_X (OUP_N / 16)        // 64 o-tiles
#define NBLK_Y (VIS_N / 256)       // 16 s-supertiles (256 s rows / block)
#define NBLOCKS (NBLK_X * NBLK_Y)  // 1024

// ---------------- zero output (atomic fallback only; d_out is poisoned) -------------
__global__ void rbm_zero_out(float* out) {
    if (threadIdx.x < 2) out[threadIdx.x] = 0.0f;
}

// ------- convert interleaved complex64 weights -> de-interleaved f16 planes --------
__global__ void rbm_cvt_w(const float* __restrict__ w,
                          _Float16* __restrict__ wr,
                          _Float16* __restrict__ wi) {
    int t = blockIdx.x * blockDim.x + threadIdx.x;     // each thread: 2 complex
    float4 f = ((const float4*)w)[t];
    wr[2 * t]     = (_Float16)f.x;
    wi[2 * t]     = (_Float16)f.y;
    wr[2 * t + 1] = (_Float16)f.z;
    wi[2 * t + 1] = (_Float16)f.w;
}

// ---------------- main GEMM + complex log-cosh, per-block partial sums --------------
// Block: 256 threads = 8 wave32. Block covers o-tile [o0,o0+16) x 256 s rows.
// Each wave owns TWO 16x16 (s x o) tiles at s0 and s0+16: consecutive s-tiles share
// half their A fragment (circulant structure), so A costs 3 segments for 4 WMMAs.
template <bool USE_WS, bool ATOMIC_OUT>
__global__ void __launch_bounds__(256)
rbm_main(const float* __restrict__ vis,
         const float* __restrict__ wcpx,      // interleaved complex (fallback path)
         const _Float16* __restrict__ wr16,   // [OUP_N][INP_N] f16 (USE_WS path)
         const _Float16* __restrict__ wi16,
         float* __restrict__ outbuf) {        // partials[2*NBLOCKS] or d_out (atomic)
    __shared__ _Float16 vs[VIS_N + 16];       // padded circular sign vector
    __shared__ float red[16];                 // per-wave partial (r,i) pairs

    const int tid = threadIdx.x;
    #pragma unroll
    for (int j = 0; j < 16; ++j) {
        int i = tid * 16 + j;
        vs[i] = (_Float16)(2.0f * vis[i] - 1.0f);    // {0,1} -> {-1,+1}
    }
    __syncthreads();
    if (tid < 16) vs[VIS_N + tid] = vs[tid];         // wrap padding
    __syncthreads();

    const int lane = tid & 31;
    const int wave = tid >> 5;
    const int hh   = lane >> 4;       // lane half -> K sub-block per ISA layout
    const int r    = lane & 15;       // row (A) / column (B,C)
    const int s0   = blockIdx.y * 256 + wave * 32;   // first s-tile of this wave
    const int o0   = blockIdx.x * 16;

    v8f aR0 = {}, aI0 = {};           // tile at s0
    v8f aR1 = {}, aI1 = {};           // tile at s0+16

    const int brow = (o0 + r) * INP_N + hh * 8;      // per-lane weight-row offset

    for (int k0 = 0; k0 < INP_N; k0 += 32) {
        // ---- A fragments from LDS. Tile S: e<8 -> vs[(S+r+k0+hh*8)+e],
        // e>=8 -> +16. Tiles s0 / s0+16 share the middle segment.
        int q  = s0 + r + hh * 8 + k0;
        int p0 = q & (VIS_N - 1);
        int p1 = (q + 16) & (VIS_N - 1);
        int p2 = (q + 32) & (VIS_N - 1);
        v16h a0, a1;
        #pragma unroll
        for (int e = 0; e < 8; ++e) {
            _Float16 m = vs[p1 + e];            // shared segment (CSE'd)
            a0[e]     = vs[p0 + e];
            a0[8 + e] = m;
            a1[e]     = m;
            a1[8 + e] = vs[p2 + e];
        }

        // ---- B fragments (32x16 f16): element e<8 -> K = hh*8+e, e>=8 -> K = 16+hh*8+(e-8), N = r
        v16h br, bi;
        if (USE_WS) {
            const _Float16* pr = wr16 + brow + k0;
            const _Float16* pi = wi16 + brow + k0;
            v8h r0 = *(const v8h*)(pr);
            v8h r1 = *(const v8h*)(pr + 16);
            v8h i0 = *(const v8h*)(pi);
            v8h i1 = *(const v8h*)(pi + 16);
            #pragma unroll
            for (int e = 0; e < 8; ++e) {
                br[e] = r0[e]; br[8 + e] = r1[e];
                bi[e] = i0[e]; bi[8 + e] = i1[e];
            }
        } else {
            const float* pw = wcpx + 2 * (size_t)(brow + k0);
            #pragma unroll
            for (int e = 0; e < 16; e += 2) {
                v4f f = *(const v4f*)(pw + 2 * (e + (e & 8)));  // (re,im,re,im)
                br[e]     = (_Float16)f[0];
                bi[e]     = (_Float16)f[1];
                br[e + 1] = (_Float16)f[2];
                bi[e + 1] = (_Float16)f[3];
            }
        }

        aR0 = __builtin_amdgcn_wmma_f32_16x16x32_f16(false, a0, false, br,
                                                     (short)0, aR0, false, false);
        aI0 = __builtin_amdgcn_wmma_f32_16x16x32_f16(false, a0, false, bi,
                                                     (short)0, aI0, false, false);
        aR1 = __builtin_amdgcn_wmma_f32_16x16x32_f16(false, a1, false, br,
                                                     (short)0, aR1, false, false);
        aI1 = __builtin_amdgcn_wmma_f32_16x16x32_f16(false, a1, false, bi,
                                                     (short)0, aI1, false, false);
    }

    // ---- complex log(cosh(x+iy)). D layout: D[v] on lane -> M = hh*8+v, N = r.
    float sr = 0.0f, si = 0.0f;
    #pragma unroll
    for (int t = 0; t < 2; ++t) {
        const v8f& xr = t ? aR1 : aR0;
        const v8f& xi = t ? aI1 : aI0;
        #pragma unroll
        for (int vv = 0; vv < 8; ++vv) {
            int s = s0 + t * 16 + hh * 8 + vv;
            if (s < S_TOTAL) {                  // drop padded shift s = 4095
                float x = xr[vv];
                float y = xi[vv];
                float ex  = __expf(x);
                float emx = __expf(-x);
                float ch = 0.5f * (ex + emx);
                float sh = 0.5f * (ex - emx);
                float sy, cy;
                __sincosf(y, &sy, &cy);
                float cre = ch * cy;            // Re cosh(z)
                float cim = sh * sy;            // Im cosh(z)
                sr += 0.5f * __logf(cre * cre + cim * cim);
                si += atan2f(cim, cre);
            }
        }
    }

    #pragma unroll
    for (int off = 16; off > 0; off >>= 1) {
        sr += __shfl_xor(sr, off, 32);
        si += __shfl_xor(si, off, 32);
    }
    if (lane == 0) { red[wave * 2] = sr; red[wave * 2 + 1] = si; }
    __syncthreads();

    if (tid == 0) {
        float pr = 0.0f, pi = 0.0f;
        #pragma unroll
        for (int wv = 0; wv < 8; ++wv) { pr += red[wv * 2]; pi += red[wv * 2 + 1]; }
        if (ATOMIC_OUT) {
            atomicAdd(&outbuf[0], pr);
            atomicAdd(&outbuf[1], pi);
        } else {
            int bid = blockIdx.y * gridDim.x + blockIdx.x;
            outbuf[2 * bid]     = pr;
            outbuf[2 * bid + 1] = pi;
        }
    }
}

// ---------------- deterministic final reduction over block partials ----------------
__global__ void __launch_bounds__(256)
rbm_finalize(const float* __restrict__ partials, float* __restrict__ out) {
    __shared__ float pr[8], pi[8];
    float sr = 0.0f, si = 0.0f;
    for (int i = threadIdx.x; i < NBLOCKS; i += 256) {
        sr += partials[2 * i];
        si += partials[2 * i + 1];
    }
    #pragma unroll
    for (int off = 16; off > 0; off >>= 1) {
        sr += __shfl_xor(sr, off, 32);
        si += __shfl_xor(si, off, 32);
    }
    int lane = threadIdx.x & 31;
    int wave = threadIdx.x >> 5;
    if (lane == 0) { pr[wave] = sr; pi[wave] = si; }
    __syncthreads();
    if (threadIdx.x == 0) {
        float a = 0.0f, b = 0.0f;
        #pragma unroll
        for (int wv = 0; wv < 8; ++wv) { a += pr[wv]; b += pi[wv]; }
        out[0] = a;
        out[1] = b;
    }
}

extern "C" void kernel_launch(void* const* d_in, const int* in_sizes, int n_in,
                              void* d_out, int out_size, void* d_ws, size_t ws_size,
                              hipStream_t stream) {
    (void)in_sizes; (void)n_in; (void)out_size;
    const float* vis = (const float*)d_in[0];
    const float* w   = (const float*)d_in[1];   // complex64 interleaved (re,im)
    float* out = (float*)d_out;

    const size_t planeElems = (size_t)OUP_N * INP_N;
    const size_t w16Bytes   = 2 * planeElems * sizeof(_Float16);   // 16 MB
    const size_t partBytes  = (size_t)NBLOCKS * 2 * sizeof(float); // 8 KB
    dim3 grid(NBLK_X, NBLK_Y);                                     // 64 x 16

    if (d_ws != nullptr && ws_size >= w16Bytes + partBytes) {
        _Float16* wr16  = (_Float16*)d_ws;
        _Float16* wi16  = wr16 + planeElems;
        float* partials = (float*)((char*)d_ws + w16Bytes);
        int t = (int)(planeElems / 2);                  // threads, 2 complex each
        rbm_cvt_w<<<t / 256, 256, 0, stream>>>(w, wr16, wi16);
        rbm_main<true, false><<<grid, 256, 0, stream>>>(vis, w, wr16, wi16, partials);
        rbm_finalize<<<1, 256, 0, stream>>>(partials, out);
    } else if (d_ws != nullptr && ws_size >= partBytes) {
        float* partials = (float*)d_ws;
        rbm_main<false, false><<<grid, 256, 0, stream>>>(vis, w, nullptr, nullptr, partials);
        rbm_finalize<<<1, 256, 0, stream>>>(partials, out);
    } else {
        rbm_zero_out<<<1, 32, 0, stream>>>(out);
        rbm_main<false, true><<<grid, 256, 0, stream>>>(vis, w, nullptr, nullptr, out);
    }
}
